// BatchGRU_5446018531351
// MI455X (gfx1250) — matrix operand order repaired and checked by reading, not compiled, not run on hardware
//
#include <hip/hip_runtime.h>
#include <hip/hip_bf16.h>
#include <stdint.h>

typedef __attribute__((ext_vector_type(16))) __bf16 v16bf;
typedef __attribute__((ext_vector_type(8)))  __bf16 v8bf;
typedef __attribute__((ext_vector_type(8)))  float  v8f;
typedef __attribute__((ext_vector_type(4)))  unsigned int v4u;
typedef __attribute__((ext_vector_type(8)))  int    v8i;
typedef __attribute__((ext_vector_type(4)))  int    v4i;

#define B_   2048
#define H_   300
#define L_   128
#define HP_  320   // K padded to 10 x 32
#define GP_  304   // per-gate column pad (19 tiles of 16)
#define NP_  912   // 3 * GP_
#define NT_  19    // 16-wide column tiles per gate

// ---------------------------------------------------------------- utilities

__device__ __forceinline__ v8f wmma_bf16(v16bf a, v16bf b, v8f c) {
  // D = A(16x32 bf16) * B(32x16 bf16) + C(16x16 f32)
  return __builtin_amdgcn_wmma_f32_16x16x32_bf16(false, a, false, b, (short)0, c,
                                                 false, false);
}

// A-operand (16x32 bf16), contiguous K order: lane<16 holds K {0..15} of row lane,
// lane>=16 holds K {16..31} of row lane-16.
__device__ __forceinline__ v16bf load_A_lds(const __bf16* base, int k0, int lane) {
  const int r = lane & 15, half = lane >> 4;
  const v8bf lo = *(const v8bf*)(base + r * HP_ + k0 + half * 16);
  const v8bf hi = *(const v8bf*)(base + r * HP_ + k0 + half * 16 + 8);
  v16bf out;
#pragma unroll
  for (int i = 0; i < 8; ++i) { out[i] = lo[i]; out[8 + i] = hi[i]; }
  return out;
}

// ---------------------------------------------------------------- kernel 0: scan

__global__ void scan_kernel(const int* __restrict__ a, int* __restrict__ starts) {
  __shared__ int s[B_];
  const int t = threadIdx.x;                 // blockDim = 1024
  s[t] = a[t]; s[t + 1024] = a[t + 1024];
  __syncthreads();
  for (int off = 1; off < B_; off <<= 1) {
    int v0 = (t        >= off) ? s[t        - off] : 0;
    int v1 = (t + 1024 >= off) ? s[t + 1024 - off] : 0;
    __syncthreads();
    s[t] += v0; s[t + 1024] += v1;
    __syncthreads();
  }
  starts[t]        = s[t]        - a[t];
  starts[t + 1024] = s[t + 1024] - a[t + 1024];
}

// ------------------------------------------------- kernel 1: bf16 weight tiles

__global__ void wt_kernel(const float* __restrict__ wih_f, const float* __restrict__ whh_f,
                          const float* __restrict__ wih_b, const float* __restrict__ whh_b,
                          __bf16* __restrict__ wihT, __bf16* __restrict__ whhT) {
  const int total = 2 * HP_ * NP_;
  int idx = blockIdx.x * blockDim.x + threadIdx.x;
  if (idx >= total) return;
  const int d = idx / (HP_ * NP_);
  const int rem = idx % (HP_ * NP_);
  const int n = rem / HP_;
  const int k = rem % HP_;
  const int g = n / GP_, e = n % GP_;
  const float* wi = d ? wih_b : wih_f;
  const float* wh = d ? whh_b : whh_f;
  float vi = 0.f, vh = 0.f;
  if (k < H_ && e < H_) {
    vi = wi[(size_t)(g * H_ + e) * H_ + k];
    vh = wh[(size_t)(g * H_ + e) * H_ + k];
  }
  wihT[idx] = (__bf16)vi;
  whhT[idx] = (__bf16)vh;
}

// ---------------------------------- kernel 2: h0 (segment max) + padded message

__global__ void prep_kernel(const float* __restrict__ node, const int* __restrict__ a_scope,
                            const int* __restrict__ starts, const float* __restrict__ bias,
                            float* __restrict__ h0g, __bf16* __restrict__ msgp) {
  const int b = blockIdx.x;
  const int start = starts[b];
  const int len = a_scope[b];
  for (int e = threadIdx.x; e < HP_; e += blockDim.x) {
    float m = 0.f;
    if (e < H_) {
      m = node[(size_t)start * H_ + e];
      for (int l = 1; l < len; ++l)
        m = fmaxf(m, node[(size_t)(start + l) * H_ + e]);
    }
    h0g[(size_t)b * HP_ + e] = m;
  }
  for (int idx = threadIdx.x; idx < L_ * HP_; idx += blockDim.x) {
    const int l = idx / HP_, e = idx % HP_;
    float v = 0.f;
    if (l < len && e < H_)
      v = fmaxf(node[(size_t)(start + l) * H_ + e] + bias[e], 0.f);
    msgp[((size_t)b * L_ + l) * HP_ + e] = (__bf16)v;
  }
}

// ----------------------------------------------------- kernel 3: fused GRU scan

__global__ __launch_bounds__(320) void gru_kernel(
    const __bf16* __restrict__ msgp, const float* __restrict__ h0g,
    const __bf16* __restrict__ wihT2, const __bf16* __restrict__ whhT2,
    const float* __restrict__ bihf, const float* __restrict__ bhhf,
    const float* __restrict__ bihb, const float* __restrict__ bhhb,
    const int* __restrict__ a_scope, const int* __restrict__ starts,
    float* __restrict__ out) {
  __shared__ __bf16 Msg[16][HP_];   // step's message tile (TDM target)
  __shared__ __bf16 Hbf[16][HP_];   // current hidden state, bf16 A-operand

  const int tid  = threadIdx.x;
  const int lane = tid & 31;
  const int wv   = tid >> 5;        // 10 waves
  const int ln   = lane & 15;
  const int half = lane >> 4;
  const int dir  = blockIdx.y;
  const int b0   = blockIdx.x * 16;

  const __bf16* wih = wihT2 + (size_t)dir * HP_ * NP_;
  const __bf16* whh = whhT2 + (size_t)dir * HP_ * NP_;
  const float* bih = dir ? bihb : bihf;
  const float* bhh = dir ? bhhb : bhhf;

  for (int i = tid; i < 16 * HP_; i += 320) {
    ((__bf16*)Hbf)[i] = (__bf16)0.f;   // pad cols 304..319 stay zero forever
    ((__bf16*)Msg)[i] = (__bf16)0.f;
  }

  int st8[8], l8[8];
#pragma unroll
  for (int j = 0; j < 8; ++j) {
    const int bb = b0 + half * 8 + j;
    st8[j] = starts[bb];
    l8[j]  = a_scope[bb];
  }

  // Each wave owns 2 column-tile slots; a slot covers the SAME 16 hidden
  // columns across all three gates, so the gate combine stays in registers.
  int  tl[2], col[2];  bool act[2];
  float bR[2], bZ[2], bNx[2], bNh[2];
  float hreg[2][8];
#pragma unroll
  for (int s = 0; s < 2; ++s) {
    const int t = wv + s * 10;
    act[s] = (t < NT_);
    tl[s]  = act[s] ? t : 0;
    col[s] = tl[s] * 16 + ln;
    const int cc = (col[s] < H_) ? col[s] : 0;
    bR[s]  = bih[cc]          + bhh[cc];
    bZ[s]  = bih[H_ + cc]     + bhh[H_ + cc];
    bNx[s] = bih[2 * H_ + cc];
    bNh[s] = bhh[2 * H_ + cc];
#pragma unroll
    for (int j = 0; j < 8; ++j)
      hreg[s][j] = h0g[(size_t)(b0 + half * 8 + j) * HP_ + col[s]];  // pad cols give 0
  }
  __syncthreads();

  for (int t = 0; t < L_; ++t) {
    const int te = dir ? (L_ - 1 - t) : t;

    // (a) publish current h as bf16 (recurrent A-operand)
#pragma unroll
    for (int s = 0; s < 2; ++s) {
      if (act[s]) {
#pragma unroll
        for (int j = 0; j < 8; ++j)
          Hbf[half * 8 + j][col[s]] = (__bf16)hreg[s][j];
      }
    }

    // (b) TDM: waves 0..7 each DMA 2 rows (2 x 320 bf16) of this step's message
    if (wv < 8) {
      const int r0 = wv * 2;
      const unsigned long long ga =
          (unsigned long long)(uintptr_t)(msgp + ((size_t)(b0 + r0) * L_ + te) * HP_);
      const unsigned lds = (unsigned)(uintptr_t)&Msg[r0][0];
      v4u g0;
      g0[0] = 1u;                                              // count=1, load
      g0[1] = lds;                                             // lds_addr
      g0[2] = (unsigned)(ga & 0xffffffffu);                    // global_addr lo
      g0[3] = (unsigned)((ga >> 32) & 0x01ffffffu) | (2u << 30); // addr hi | type=2
      v8i g1;
      g1[0] = 1 << 16;            // data_size = 2 bytes
      g1[1] = HP_ << 16;          // tensor_dim0 = 320 (bits 63:48)
      g1[2] = 2 << 16;            // tensor_dim1 = 2   (bits 95:80)
      g1[3] = HP_ << 16;          // tile_dim0 = 320   (bits 127:112)
      g1[4] = 2;                  // tile_dim1 = 2, tile_dim2 = 0
      g1[5] = L_ * HP_;           // tensor_dim0_stride = 40960 elements
      g1[6] = 0;
      g1[7] = 0;
      v4i zz4; zz4[0] = zz4[1] = zz4[2] = zz4[3] = 0;
      v8i zz8;
#pragma unroll
      for (int i = 0; i < 8; ++i) zz8[i] = 0;
      __builtin_amdgcn_tensor_load_to_lds(g0, g1, zz4, zz4, zz8, 0);
      __builtin_amdgcn_s_wait_tensorcnt(0);
    }
    __syncthreads();

    // (c) GEMMs + gate combine, fully register resident per slot
#pragma unroll
    for (int s = 0; s < 2; ++s) {
      const int nb = tl[s] * 16;
      v8f accR  = {0.f,0.f,0.f,0.f,0.f,0.f,0.f,0.f};
      v8f accZ  = {0.f,0.f,0.f,0.f,0.f,0.f,0.f,0.f};
      v8f accXN = {0.f,0.f,0.f,0.f,0.f,0.f,0.f,0.f};
      v8f accHN = {0.f,0.f,0.f,0.f,0.f,0.f,0.f,0.f};
#pragma unroll 2
      for (int kk = 0; kk < HP_ / 32; ++kk) {
        const int k0 = kk * 32;
        const v16bf am = load_A_lds(&Msg[0][0], k0, lane);
        const v16bf ah = load_A_lds(&Hbf[0][0], k0, lane);
        const __bf16* wkI = wih + (size_t)(nb + ln) * HP_ + k0 + half * 16;
        const __bf16* wkH = whh + (size_t)(nb + ln) * HP_ + k0 + half * 16;
        const v16bf bIR = *(const v16bf*)(wkI);
        const v16bf bIZ = *(const v16bf*)(wkI + (size_t)GP_ * HP_);
        const v16bf bIN = *(const v16bf*)(wkI + (size_t)2 * GP_ * HP_);
        const v16bf bHR = *(const v16bf*)(wkH);
        const v16bf bHZ = *(const v16bf*)(wkH + (size_t)GP_ * HP_);
        const v16bf bHN = *(const v16bf*)(wkH + (size_t)2 * GP_ * HP_);
        accR  = wmma_bf16(am, bIR, accR);
        accR  = wmma_bf16(ah, bHR, accR);   // r-gate: x-part + h-part share acc
        accZ  = wmma_bf16(am, bIZ, accZ);
        accZ  = wmma_bf16(ah, bHZ, accZ);
        accXN = wmma_bf16(am, bIN, accXN);  // n-gate parts kept separate (r * hn)
        accHN = wmma_bf16(ah, bHN, accHN);
      }
      const bool cok = (col[s] < H_);
#pragma unroll
      for (int j = 0; j < 8; ++j) {
        const float r  = 1.f / (1.f + __expf(-(accR[j] + bR[s])));
        const float z  = 1.f / (1.f + __expf(-(accZ[j] + bZ[s])));
        const float n  = tanhf(accXN[j] + bNx[s] + r * (accHN[j] + bNh[s]));
        float hn = (1.f - z) * n + z * hreg[s][j];
        hn = cok ? hn : 0.f;
        hreg[s][j] = hn;
        if (act[s] && cok && te < l8[j]) {
          out[(size_t)(st8[j] + te) * (2 * H_) + (size_t)dir * H_ + col[s]] = hn;
        }
      }
    }
    __syncthreads();   // protect Msg/Hbf before next step rewrites them
  }
}

// ---------------------------------------------------------------- entry point

extern "C" void kernel_launch(void* const* d_in, const int* in_sizes, int n_in,
                              void* d_out, int out_size, void* d_ws, size_t ws_size,
                              hipStream_t stream) {
  (void)in_sizes; (void)n_in; (void)out_size; (void)ws_size;
  const float* node    = (const float*)d_in[0];
  const int*   a_scope = (const int*)d_in[1];
  const float* bias    = (const float*)d_in[3];
  const float* wih_f   = (const float*)d_in[4];
  const float* whh_f   = (const float*)d_in[5];
  const float* bih_f   = (const float*)d_in[6];
  const float* bhh_f   = (const float*)d_in[7];
  const float* wih_b   = (const float*)d_in[8];
  const float* whh_b   = (const float*)d_in[9];
  const float* bih_b   = (const float*)d_in[10];
  const float* bhh_b   = (const float*)d_in[11];
  float* out = (float*)d_out;

  char* ws = (char*)d_ws;
  size_t off = 0;
  auto alloc = [&](size_t bytes) -> void* {
    void* p = ws + off;
    off += (bytes + 255) & ~(size_t)255;
    return p;
  };
  int*    starts = (int*)   alloc((size_t)B_ * 4);
  float*  h0g    = (float*) alloc((size_t)B_ * HP_ * 4);
  __bf16* wihT   = (__bf16*)alloc((size_t)2 * HP_ * NP_ * 2);
  __bf16* whhT   = (__bf16*)alloc((size_t)2 * HP_ * NP_ * 2);
  __bf16* msgp   = (__bf16*)alloc((size_t)B_ * L_ * HP_ * 2);

  scan_kernel<<<1, 1024, 0, stream>>>(a_scope, starts);
  {
    const int tot = 2 * HP_ * NP_;
    wt_kernel<<<(tot + 255) / 256, 256, 0, stream>>>(wih_f, whh_f, wih_b, whh_b,
                                                     wihT, whhT);
  }
  prep_kernel<<<B_, 256, 0, stream>>>(node, a_scope, starts, bias, h0g, msgp);
  gru_kernel<<<dim3(B_ / 16, 2), 320, 0, stream>>>(msgp, h0g, wihT, whhT,
                                                   bih_f, bhh_f, bih_b, bhh_b,
                                                   a_scope, starts, out);
}
